// PatchSampleLayer_29025388987060
// MI455X (gfx1250) — matrix operand order repaired and checked by reading, not compiled, not run on hardware
//
#include <hip/hip_runtime.h>
#include <hip/hip_bf16.h>

typedef __attribute__((ext_vector_type(16))) _Float16     v16h;
typedef __attribute__((ext_vector_type(8)))  float        v8f;
typedef __attribute__((ext_vector_type(4)))  unsigned int v4u;
typedef __attribute__((ext_vector_type(8)))  int          v8i;
typedef __attribute__((ext_vector_type(4)))  int          v4i;

#define B_     2
#define N_     256
#define NB     512          // N_*B_
#define IN_CH  256
#define HID    64
#define C_HM   62
#define HIN    180
#define WIN    180
#define HS     40
#define WS     40
#define PO     38           // conv output spatial
#define M_OUT  (PO*PO)      // 1444 output pixels per image
#define K_DIM  (C_HM*9)     // 558
#define NCHUNK 18           // ceil(558/32)
#define MBLOCKS 23          // ceil(ceil(1444/16)/4)

// LDS layout (halves) inside the single shared array of the conv kernel:
//   [0, 12400)      : activation tile, 62ch x 5rows x 40cols ([c][ry][x], c-stride 200)
//   [12400, 12656)  : zero pad slot (targets for padded-K gathers)
//   [12656, 13232)  : im2col offset table, 18 chunks x 2 halves x 16 e  (u16)
#define TILE_H   12400
#define ZPAD_OFF 12400
#define ZPAD_H   256
#define TBL_OFF  12656
#define TBL_H    576
#define SMEM_H   (TBL_OFF + TBL_H)

// ---------------------------------------------------------------- MLP -> theta
__global__ void mlp_kernel(const float* __restrict__ q,
                           const float* __restrict__ w1, const float* __restrict__ b1,
                           const float* __restrict__ w2, const float* __restrict__ b2,
                           const float* __restrict__ w3, const float* __restrict__ b3,
                           float* __restrict__ theta) {
  __shared__ float sq[IN_CH];
  __shared__ float h1[HID];
  __shared__ float h2[HID];
  int bq = blockIdx.x;              // b*N_ + n
  int t  = threadIdx.x;             // 0..63
  const float* qp = q + (size_t)bq * IN_CH;
  for (int i = t; i < IN_CH; i += HID) sq[i] = qp[i];
  __syncthreads();
  float acc = b1[t];
  const float* wr = w1 + (size_t)t * IN_CH;
  for (int i = 0; i < IN_CH; ++i) acc += sq[i] * wr[i];
  h1[t] = fmaxf(acc, 0.f);
  __syncthreads();
  acc = b2[t];
  const float* wr2 = w2 + (size_t)t * HID;
  for (int i = 0; i < HID; ++i) acc += h1[i] * wr2[i];
  h2[t] = fmaxf(acc, 0.f);
  __syncthreads();
  if (t < 6) {
    float a = b3[t];
    const float* wr3 = w3 + (size_t)t * HID;
    for (int i = 0; i < HID; ++i) a += h2[i] * wr3[i];
    theta[bq * 6 + t] = a;
  }
}

// ------------------------------------------- affine grid + bilinear sampling
// Writes f16 activations X[img][c][h][w], img = n*B_ + b
__global__ void sample_kernel(const float* __restrict__ hm,
                              const float* __restrict__ theta,
                              _Float16* __restrict__ X) {
  int img = blockIdx.x;
  int b = img & (B_ - 1);
  int n = img >> 1;
  const float* th = theta + ((size_t)b * N_ + n) * 6;
  float t0 = th[0], t1 = th[1], t2 = th[2], t3 = th[3], t4 = th[4], t5 = th[5];
  for (int pix = threadIdx.x; pix < HS * WS; pix += blockDim.x) {
    int h = pix / WS, w = pix - (pix / WS) * WS;
    float gx = (2.f * (float)w + 1.f) / (float)WS - 1.f;
    float gy = (2.f * (float)h + 1.f) / (float)HS - 1.f;
    float x = t0 * gx + t1 * gy + t2;
    float y = t3 * gx + t4 * gy + t5;
    float ix = ((x + 1.f) * (float)WIN - 1.f) * 0.5f;
    float iy = ((y + 1.f) * (float)HIN - 1.f) * 0.5f;
    float x0f = floorf(ix), y0f = floorf(iy);
    float wx1 = ix - x0f, wy1 = iy - y0f;
    int x0 = (int)x0f, y0 = (int)y0f;
    int xi[4] = { x0, x0 + 1, x0,     x0 + 1 };
    int yi[4] = { y0, y0,     y0 + 1, y0 + 1 };
    float wt[4] = { (1.f - wx1) * (1.f - wy1), wx1 * (1.f - wy1),
                    (1.f - wx1) * wy1,         wx1 * wy1 };
    int off[4];
    #pragma unroll
    for (int k = 0; k < 4; ++k) {
      bool in = (xi[k] >= 0) && (xi[k] < WIN) && (yi[k] >= 0) && (yi[k] < HIN);
      int xc = xi[k] < 0 ? 0 : (xi[k] > WIN - 1 ? WIN - 1 : xi[k]);
      int yc = yi[k] < 0 ? 0 : (yi[k] > HIN - 1 ? HIN - 1 : yi[k]);
      off[k] = yc * WIN + xc;
      if (!in) wt[k] = 0.f;
    }
    const float* base = hm + (size_t)b * C_HM * HIN * WIN;
    _Float16* xo = X + (size_t)img * C_HM * (HS * WS) + pix;
    for (int c = 0; c < C_HM; ++c) {
      const float* bc = base + (size_t)c * HIN * WIN;
      float v = wt[0] * bc[off[0]] + wt[1] * bc[off[1]]
              + wt[2] * bc[off[2]] + wt[3] * bc[off[3]];
      xo[(size_t)c * (HS * WS)] = (_Float16)v;
    }
  }
}

// -------- repack conv weights into per-(chunk, ntile, lane) v16h B-fragments
// 16-bit B 32x16 order used here (contiguous k): N = lane&15, K = e + 16*(lane>>4)
__global__ void wrepack_kernel(const float* __restrict__ w, _Float16* __restrict__ Wf) {
  int idx = blockIdx.x * blockDim.x + threadIdx.x;   // [ch][nt][lane][e]
  if (idx >= NCHUNK * 4 * 32 * 16) return;
  int e    = idx & 15;
  int lane = (idx >> 4) & 31;
  int nt   = (idx >> 9) & 3;
  int ch   = idx >> 11;
  int k  = ch * 32 + e + 16 * (lane >> 4);
  int nn = nt * 16 + (lane & 15);
  float v = 0.f;
  if (k < K_DIM) {
    int c = k / 9; int rs = k - c * 9; int r = rs / 3; int s = rs - r * 3;
    v = w[(((size_t)nn * C_HM + c) * 3 + r) * 3 + s];
  }
  Wf[idx] = (_Float16)v;
}

// --------------------------------------------- 3x3 conv as implicit-GEMM WMMA
__global__ __launch_bounds__(128)
void conv_wmma_kernel(const _Float16* __restrict__ X,
                      const _Float16* __restrict__ Wf,
                      const float* __restrict__ cb,
                      _Float16* __restrict__ Y) {
  // Single shared object -> LDS offset 0 (used as TDM lds_addr).
  __shared__ _Float16 smem[SMEM_H];
  unsigned short* Tarr = (unsigned short*)(void*)(smem + TBL_OFF);

  int img = blockIdx.x;
  int mb  = blockIdx.y;
  int m0  = mb * 64;
  int p_min = m0 / PO;
  int t = threadIdx.x;

  // ---- build im2col offset table + zero pad (all threads) -----------------
  for (int idx = t; idx < TBL_H; idx += 128) {
    int e    = idx & 15;
    int half = (idx >> 4) & 1;
    int ch   = idx >> 5;
    int kk = ch * 32 + e + 16 * half;
    unsigned short v = (unsigned short)ZPAD_OFF;
    if (kk < K_DIM) {
      int c = kk / 9; int rs = kk - c * 9; int r = rs / 3; int s = rs - r * 3;
      v = (unsigned short)(c * 200 + r * 40 + s);
    }
    Tarr[idx] = v;
  }
  for (int idx = t; idx < ZPAD_H; idx += 128) smem[ZPAD_OFF + idx] = (_Float16)0.f;

  // ---- stage activation tile Global -> LDS ---------------------------------
  const _Float16* Xi = X + (size_t)img * C_HM * HS * WS;
#if __has_builtin(__builtin_amdgcn_tensor_load_to_lds)
  if (t < 32) {                                  // wave 0 issues the TDM DMA
    unsigned long long ga =
        (unsigned long long)(uintptr_t)(Xi + (size_t)p_min * WS);
    unsigned td1 = (unsigned)(HS - p_min);       // rows left; OOB rows -> zeros
    v4u g0 = { 1u,                               // count=1, user mode
               0u,                               // lds_addr = 0 (tile at smem[0])
               (unsigned)(ga & 0xFFFFFFFFull),
               (unsigned)((ga >> 32) & 0x1FFFFFFull) | (2u << 30) }; // type=2
    v8i g1 = { (int)(1u << 16),                  // data_size=1 (2 bytes)
               (int)(((unsigned)WS & 0xFFFFu) << 16),   // tensor_dim0 = 40
               (int)((td1 & 0xFFFFu) << 16),            // tensor_dim1 = 40-p_min
               (int)(((unsigned)WS) << 16),             // tile_dim0 = 40
               (int)(5u | (62u << 16)),                 // tile_dim1=5, tile_dim2=62
               (int)WS,                                 // dim0 stride = 40
               (int)((unsigned)(HS * WS) << 16),        // dim1 stride = 1600
               0 };
    v4i g2 = { (int)C_HM, 0, 0, 0 };            // tensor_dim2 = 62
    v4i g3 = { 0, 0, 0, 0 };
#if defined(__clang_major__) && (__clang_major__ >= 23)
    v8i gz = { 0, 0, 0, 0, 0, 0, 0, 0 };
    __builtin_amdgcn_tensor_load_to_lds(g0, g1, g2, g3, gz, 0);
#else
    __builtin_amdgcn_tensor_load_to_lds(g0, g1, g2, g3, 0);
#endif
    __builtin_amdgcn_s_wait_tensorcnt((short)0);
  }
#else
  for (int i = t; i < TILE_H; i += 128) {
    int c = i / 200; int rem = i - c * 200; int ry = rem / 40; int x = rem - ry * 40;
    int yy = p_min + ry;
    smem[i] = (yy < HS) ? Xi[((size_t)c * HS + yy) * WS + x] : (_Float16)0.f;
  }
#endif
  __syncthreads();

  // ---- WMMA main loop ------------------------------------------------------
  int wave = t >> 5;
  int lane = t & 31;
  int mt = mb * 4 + wave;
  int m_base = mt * 16;
  if (m_base >= M_OUT) return;                   // wave-uniform

  int half = lane >> 4;
  int m = m_base + (lane & 15);
  int mc = m < M_OUT ? m : M_OUT - 1;            // clamped rows never stored
  int p = mc / PO, qq = mc - (mc / PO) * PO;
  int base = (p - p_min) * 40 + qq;              // 0..117

  v8f acc[4] = {};
  const v16h* Wv = (const v16h*)Wf;
  for (int ch = 0; ch < NCHUNK; ++ch) {
    const unsigned short* Trow = Tarr + (ch * 2 + half) * 16;
    v16h a;
    #pragma unroll
    for (int e = 0; e < 16; ++e) a[e] = smem[(int)Trow[e] + base];
    #pragma unroll
    for (int nt = 0; nt < 4; ++nt) {
      v16h bf = Wv[(ch * 4 + nt) * 32 + lane];
      acc[nt] = __builtin_amdgcn_wmma_f32_16x16x32_f16(
          /*neg_a=*/false, a, /*neg_b=*/false, bf,
          /*c_mod=*/(short)0, acc[nt], /*reuse_a=*/false, /*reuse_b=*/false);
    }
  }

  // D layout: N = lane&15, M = r + 8*half. Add conv bias, store f16 [img][m][o]
  #pragma unroll
  for (int nt = 0; nt < 4; ++nt) {
    int ncol = nt * 16 + (lane & 15);
    float bb = cb[ncol];
    #pragma unroll
    for (int r = 0; r < 8; ++r) {
      int mrow = m_base + r + 8 * half;
      if (mrow < M_OUT)
        Y[((size_t)img * M_OUT + mrow) * HID + ncol] = (_Float16)(acc[nt][r] + bb);
    }
  }
}

// --------------------------------- BN stats: per-channel partial sum / sumsq
__global__ void bnpart_kernel(const _Float16* __restrict__ Y, float* __restrict__ part) {
  __shared__ float ss[256], sq[256];
  int blk = blockIdx.x;                 // 128 blocks
  int t = threadIdx.x;                  // 256 threads
  int o = t & 63;
  int sub = t >> 6;
  const int RPB = (NB * M_OUT) / 128;   // 5776 rows per block
  float s = 0.f, q2 = 0.f;
  for (int j = sub; j < RPB; j += 4) {
    float v = (float)Y[((size_t)blk * RPB + j) * HID + o];
    s += v; q2 += v * v;
  }
  ss[t] = s; sq[t] = q2;
  __syncthreads();
  if (t < 64) {
    float S = ss[t] + ss[t + 64] + ss[t + 128] + ss[t + 192];
    float Q = sq[t] + sq[t + 64] + sq[t + 128] + sq[t + 192];
    part[(blk * 64 + t) * 2]     = S;
    part[(blk * 64 + t) * 2 + 1] = Q;
  }
}

__global__ void bnstat_kernel(const float* __restrict__ part, float* __restrict__ stats) {
  int o = threadIdx.x;                  // 64
  float S = 0.f, Q = 0.f;
  for (int b = 0; b < 128; ++b) { S += part[(b * 64 + o) * 2]; Q += part[(b * 64 + o) * 2 + 1]; }
  float inv = 1.f / (float)(NB * M_OUT);
  float mu  = S * inv;
  float var = Q * inv - mu * mu;
  stats[o * 2]     = mu;
  stats[o * 2 + 1] = rsqrtf(var + 1e-5f);
}

// ----------------------------- BN + ReLU + global-avg-pool + final 64->2 FC
__global__ void pool_fc_kernel(const _Float16* __restrict__ Y,
                               const float* __restrict__ stats,
                               const float* __restrict__ gamma,
                               const float* __restrict__ beta,
                               const float* __restrict__ w2,
                               const float* __restrict__ b2,
                               float* __restrict__ out) {
  __shared__ float pooled[HID];
  int img = blockIdx.x;
  int o = threadIdx.x;                  // 64
  float mu = stats[o * 2], rs = stats[o * 2 + 1];
  float g = gamma[o], be = beta[o];
  float s = 0.f;
  const _Float16* yp = Y + (size_t)img * M_OUT * HID + o;
  for (int m = 0; m < M_OUT; ++m) {
    float v = (float)yp[(size_t)m * HID];
    v = (v - mu) * rs * g + be;
    s += fmaxf(v, 0.f);
  }
  pooled[o] = s / (float)M_OUT;
  __syncthreads();
  if (o < 2) {
    float a = b2[o];
    for (int i = 0; i < HID; ++i) a += pooled[i] * w2[o * HID + i];
    out[img * 2 + o] = a;
  }
}

// ---------------------------------------------------------------------------
extern "C" void kernel_launch(void* const* d_in, const int* in_sizes, int n_in,
                              void* d_out, int out_size, void* d_ws, size_t ws_size,
                              hipStream_t stream) {
  const float* query = (const float*)d_in[0];
  const float* hm    = (const float*)d_in[1];
  const float* fc1w  = (const float*)d_in[2];
  const float* fc1b  = (const float*)d_in[3];
  const float* fc2w  = (const float*)d_in[4];
  const float* fc2b  = (const float*)d_in[5];
  const float* fc3w  = (const float*)d_in[6];
  const float* fc3b  = (const float*)d_in[7];
  const float* c1w   = (const float*)d_in[8];
  const float* c1b   = (const float*)d_in[9];
  const float* gam   = (const float*)d_in[10];
  const float* bet   = (const float*)d_in[11];
  const float* c2w   = (const float*)d_in[12];
  const float* c2b   = (const float*)d_in[13];
  float* out = (float*)d_out;

  char* ws = (char*)d_ws;
  float*    theta = (float*)   (ws + 0);                       // 12,288 B
  _Float16* Wf    = (_Float16*)(ws + 12288);                   // 73,728 B
  float*    part  = (float*)   (ws + 86016);                   // 65,536 B
  float*    stats = (float*)   (ws + 151552);                  //    512 B
  _Float16* X     = (_Float16*)(ws + 152064);                  // 101,580,800 B
  _Float16* Y     = (_Float16*)(ws + 152064 + 101580800ull);   //  94,633,984 B

  mlp_kernel<<<NB, HID, 0, stream>>>(query, fc1w, fc1b, fc2w, fc2b, fc3w, fc3b, theta);
  wrepack_kernel<<<(NCHUNK * 4 * 32 * 16) / 256, 256, 0, stream>>>(c1w, Wf);
  sample_kernel<<<NB, 256, 0, stream>>>(hm, theta, X);
  dim3 cgrid(NB, MBLOCKS);
  conv_wmma_kernel<<<cgrid, 128, 0, stream>>>(X, Wf, c1b, Y);
  bnpart_kernel<<<128, 256, 0, stream>>>(Y, part);
  bnstat_kernel<<<1, 64, 0, stream>>>(part, stats);
  pool_fc_kernel<<<NB, HID, 0, stream>>>(Y, stats, gam, bet, c2w, c2b, out);
}
